// BaselineModel_32452772888974
// MI455X (gfx1250) — matrix operand order repaired and checked by reading, not compiled, not run on hardware
//
#include <hip/hip_runtime.h>
#include <hip/hip_bf16.h>

// Problem constants from the reference: B=2, L=128, H=768, R=4, T=3.
#define BB 2
#define LL 128
#define HH 768
#define RR 4
#define TT 3
#define MM (BB * LL)          // 256 GEMM rows
#define NSRC (RR * HH)        // 3072
#define NDENSE (TT * HH)      // 2304
#define HPL (HH / 32)         // 24 elements per lane (wave32)

typedef __attribute__((ext_vector_type(16))) __bf16 v16bf;
typedef __attribute__((ext_vector_type(8)))  float  v8f;

// ---------------------------------------------------------------------------
// tanh: prefer the native CDNA5 TRANS op v_tanh_f32 (gfx1250 tanh-insts);
// fall back to exp2/rcp TRANS-op emulation if the builtin is unavailable.
// ---------------------------------------------------------------------------
__device__ __forceinline__ float fast_tanh(float x) {
#if __has_builtin(__builtin_amdgcn_tanhf)
    return __builtin_amdgcn_tanhf(x);
#else
    float ax = __builtin_fabsf(x);
    float e  = __builtin_amdgcn_exp2f(ax * -2.8853900817779268f); // exp(-2|x|)
    float r  = (1.0f - e) * __builtin_amdgcn_rcpf(1.0f + e);
    return __builtin_copysignf(r, x);
#endif
}

__device__ __forceinline__ __bf16 to_bf16(float f) { return (__bf16)f; }

// ---------------------------------------------------------------------------
// GEMM with bias: out[m,n] = sum_k X[m,k] * W[n,k] + bias[n]
// X: [256, 768] fp32, W: [N, 768] fp32 (row-major, W^T used logically).
// One wave per 16x16 output tile.  fp32 emulated via bf16 hi/lo split:
//   x*w ~= xh*wh + xh*wl + xl*wh   (three v_wmma_f32_16x16x32_bf16 per K-step)
//
// Fragment layouts per CDNA5 ISA 7.12.2 (wave32):
//  A (16x32 bf16): lane row M = lane%16; elems 0..7 -> K = kb..kb+7,
//                  elems 8..15 -> K = kb+16..kb+23, kb = (lane/16)*8.
//  B (32x16 bf16): lane col N = lane%16; same K mapping as A.
//  C/D (16x16 f32): N = lane%16, M = (lane/16)*8 + i for VGPR i.
// ---------------------------------------------------------------------------
__global__ void gemm_bias_wmma(const float* __restrict__ X,
                               const float* __restrict__ W,
                               const float* __restrict__ bias,
                               float* __restrict__ out,
                               int N) {
    const int lane  = threadIdx.x & 31;
    const int wave  = threadIdx.x >> 5;
    const int wpb   = blockDim.x >> 5;
    const int wid   = blockIdx.x * wpb + wave;

    const int ntiles = N >> 4;
    const int total  = (MM >> 4) * ntiles;
    if (wid >= total) return;                 // wave-uniform exit

    const int mTile = wid / ntiles;
    const int nTile = wid % ntiles;

    const int half  = lane >> 4;              // 0 (lanes 0-15) / 1 (16-31)
    const int row   = mTile * 16 + (lane & 15);
    const int col   = nTile * 16 + (lane & 15);
    const int aKoff = half * 8;
    const int bKoff = half * 8;

    const float* __restrict__ xrow = X + (size_t)row * HH;
    const float* __restrict__ wrow = W + (size_t)col * HH;

    v8f acc = {};
    for (int kk = 0; kk < HH; kk += 32) {
        // ---- A fragment: 16 fp32 -> bf16 hi/lo -------------------------
        float af[16];
        const int k0 = kk + aKoff;
        *(float4*)(af + 0)  = *(const float4*)(xrow + k0);
        *(float4*)(af + 4)  = *(const float4*)(xrow + k0 + 4);
        *(float4*)(af + 8)  = *(const float4*)(xrow + k0 + 16);
        *(float4*)(af + 12) = *(const float4*)(xrow + k0 + 20);

        v16bf ah, al;
        #pragma unroll
        for (int e = 0; e < 16; ++e) {
            __bf16 h = to_bf16(af[e]);
            ah[e] = h;
            al[e] = to_bf16(af[e] - (float)h);
        }

        // ---- B fragment: 16 fp32 from W row, same K order as A ---------
        float wf[16];
        const int k1 = kk + bKoff;
        *(float4*)(wf + 0)  = *(const float4*)(wrow + k1);
        *(float4*)(wf + 4)  = *(const float4*)(wrow + k1 + 4);
        *(float4*)(wf + 8)  = *(const float4*)(wrow + k1 + 16);
        *(float4*)(wf + 12) = *(const float4*)(wrow + k1 + 20);

        v16bf wh, wl;
        #pragma unroll
        for (int e = 0; e < 16; ++e) {
            __bf16 h = to_bf16(wf[e]);
            wh[e] = h;
            wl[e] = to_bf16(wf[e] - (float)h);
        }

        // ---- 3-term bf16 emulation of fp32 MMA -------------------------
        acc = __builtin_amdgcn_wmma_f32_16x16x32_bf16(false, ah, false, wh,
                                                      (short)0, acc, false, false);
        acc = __builtin_amdgcn_wmma_f32_16x16x32_bf16(false, ah, false, wl,
                                                      (short)0, acc, false, false);
        acc = __builtin_amdgcn_wmma_f32_16x16x32_bf16(false, al, false, wh,
                                                      (short)0, acc, false, false);
    }

    const float bv  = bias[col];
    const int mBase = mTile * 16 + half * 8;
    float* orow = out + (size_t)mBase * N + col;
    #pragma unroll
    for (int i = 0; i < 8; ++i) {
        *orow = acc[i] + bv;
        orow += N;
    }
}

// ---------------------------------------------------------------------------
// relation_logits[b,r,s,t] = sum_h tanh(h_src[b,s,r,h] + h_tgt[b,t,r,h]) * w_out[h]
// One wave per (b,r,s); the h_src row and w_out live in registers (24 f32
// each per lane), so the inner loop is pure global-load + add + v_tanh + fma.
// ---------------------------------------------------------------------------
__global__ void relation_kernel(const float* __restrict__ h_src,
                                const float* __restrict__ h_tgt,
                                const float* __restrict__ w_out,
                                float* __restrict__ rel) {
    const int lane = threadIdx.x & 31;
    const int wid  = (int)((blockIdx.x * blockDim.x + threadIdx.x) >> 5); // 0..1023
    const int s = wid & (LL - 1);
    const int r = (wid >> 7) & (RR - 1);
    const int b = wid >> 9;

    float srcv[HPL], wv[HPL];
    const float* src = h_src + (((size_t)b * LL + s) * RR + r) * HH;
    #pragma unroll
    for (int i = 0; i < HPL; ++i) {
        srcv[i] = src[lane + 32 * i];
        wv[i]   = w_out[lane + 32 * i];
    }

    const float* tgt_base = h_tgt + ((size_t)b * LL * RR + r) * HH;
    float* orow = rel + (((size_t)b * RR + r) * LL + s) * LL;

    for (int t = 0; t < LL; ++t) {
        const float* tgt = tgt_base + (size_t)t * (RR * HH);
        float sum = 0.0f;
        #pragma unroll
        for (int i = 0; i < HPL; ++i)
            sum += fast_tanh(srcv[i] + tgt[lane + 32 * i]) * wv[i];
        #pragma unroll
        for (int off = 16; off; off >>= 1)
            sum += __shfl_xor(sum, off, 32);
        if (lane == 0)
            orow[t] = sum;
    }
}

// ---------------------------------------------------------------------------
// source_mask_logits[b,t,l] = sum_h tanh(z[b,l,t,h]) * clf_W[h] + clf_b
// One wave per output element (768 waves total).
// ---------------------------------------------------------------------------
__global__ void mask_kernel(const float* __restrict__ z,
                            const float* __restrict__ clf_W,
                            const float* __restrict__ clf_b,
                            float* __restrict__ out) {
    const int wid  = (int)((blockIdx.x * blockDim.x + threadIdx.x) >> 5);
    const int lane = threadIdx.x & 31;
    if (wid >= BB * TT * LL) return;

    const int l = wid & (LL - 1);
    const int t = (wid >> 7) % TT;
    const int b = wid / (LL * TT);

    const float* zrow = z + (((size_t)b * LL + l) * TT + t) * HH;
    float sum = 0.0f;
    #pragma unroll
    for (int i = 0; i < HPL; ++i)
        sum += fast_tanh(zrow[lane + 32 * i]) * clf_W[lane + 32 * i];
    #pragma unroll
    for (int off = 16; off; off >>= 1)
        sum += __shfl_xor(sum, off, 32);
    if (lane == 0)
        out[((size_t)b * TT + t) * LL + l] = sum + clf_b[0];
}

// ---------------------------------------------------------------------------
extern "C" void kernel_launch(void* const* d_in, const int* in_sizes, int n_in,
                              void* d_out, int out_size, void* d_ws, size_t ws_size,
                              hipStream_t stream) {
    const float* hidden  = (const float*)d_in[0];
    const float* W_src   = (const float*)d_in[1];
    const float* b_src   = (const float*)d_in[2];
    const float* W_tgt   = (const float*)d_in[3];
    const float* b_tgt   = (const float*)d_in[4];
    const float* w_out   = (const float*)d_in[5];
    const float* dense_W = (const float*)d_in[6];
    const float* dense_b = (const float*)d_in[7];
    const float* clf_W   = (const float*)d_in[8];
    const float* clf_b   = (const float*)d_in[9];

    float* out  = (float*)d_out;
    float* rel  = out;                                   // [B,R,L,L] = 131072
    float* mask = rel + (size_t)BB * RR * LL * LL;       // [B,T,L]   = 768
    float* hsrc = mask + (size_t)BB * TT * LL;           // [B,L,R,H] = 786432
    float* htgt = hsrc + (size_t)MM * NSRC;              // [B,L,R,H] = 786432

    float* z = (float*)d_ws;                             // [B,L,T,H] = 589824 f32

    const dim3 blk(128);   // 4 waves per block, 1 wave per 16x16 tile
    // src: 16*192 tiles / 4 = 768 blocks;  tgt: same;  dense: 16*144/4 = 576.
    gemm_bias_wmma<<<768, blk, 0, stream>>>(hidden, W_src,   b_src,   hsrc, NSRC);
    gemm_bias_wmma<<<768, blk, 0, stream>>>(hidden, W_tgt,   b_tgt,   htgt, NSRC);
    gemm_bias_wmma<<<576, blk, 0, stream>>>(hidden, dense_W, dense_b, z,    NDENSE);

    // 1024 waves, one per (b,r,s):  128 blocks x 256 threads.
    relation_kernel<<<(BB * RR * LL * 32) / 256, 256, 0, stream>>>(hsrc, htgt, w_out, rel);
    // 768 waves, one per (b,t,l):   96 blocks x 256 threads.
    mask_kernel<<<(BB * TT * LL * 32) / 256, 256, 0, stream>>>(z, clf_W, clf_b, mask);
}
